// Criterion_1211180777884
// MI455X (gfx1250) — compile-verified, matrix operand order checked by reading, not run on hardware
//
#include <hip/hip_runtime.h>
#include <hip/hip_bf16.h>
#include <math.h>

// ---------------------------------------------------------------------------
// CDNA5 / gfx1250: multi-scale-dice + sobel criterion.
// Double avg-pool3d == separable banded matrix (Box^2/k^2) per axis, run as
// v_wmma_f32_16x16x32_f16 GEMM tiles. M is symmetric -> column reads become
// row reads (contiguous b128). Strided operands staged via async-to-LDS.
// ---------------------------------------------------------------------------

typedef __attribute__((ext_vector_type(16))) _Float16 v16h;
typedef __attribute__((ext_vector_type(8)))  _Float16 v8h;
typedef __attribute__((ext_vector_type(8)))  float    v8f;
typedef __attribute__((ext_vector_type(4)))  float    v4f;
typedef int v4i __attribute__((vector_size(16)));      // matches builtin param type
typedef __attribute__((address_space(3))) v4i* lds_v4i_ptr;

#define NVOL   442368      // 2*24*96*96
#define INNER  221184      // 24*96*96

#ifndef __has_builtin
#define __has_builtin(x) 0
#endif
#if __has_builtin(__builtin_amdgcn_global_load_async_to_lds_b128) && \
    __has_builtin(__builtin_amdgcn_s_wait_asynccnt)
#define HAVE_ASYNC_LDS 1
#else
#define HAVE_ASYNC_LDS 0
#endif

#define SHUF16(lo, hi) \
  __builtin_shufflevector(lo, hi, 0, 1, 2, 3, 4, 5, 6, 7, 8, 9, 10, 11, 12, 13, 14, 15)

// ---------------------------------------------------------------------------
__device__ __forceinline__ float block_reduce_add(float v, float* smem) {
  int t = threadIdx.x;
  smem[t] = v;
  __syncthreads();
  for (int o = blockDim.x >> 1; o > 0; o >>= 1) {
    if (t < o) smem[t] += smem[t + o];
    __syncthreads();
  }
  float r = smem[0];
  __syncthreads();
  return r;
}

__global__ void k_write_zero(float* out) { out[0] = 0.0f; }

__global__ void k_zero(float* acc, int n) {
  int i = blockIdx.x * blockDim.x + threadIdx.x;
  if (i < n) acc[i] = 0.0f;
}

// ---------------------------------------------------------------------------
// M = Box_k^2 / k^2 (symmetric banded). Entry (c,j) = overlap count / k^2.
// ---------------------------------------------------------------------------
__global__ void k_build_m(_Float16* M, int n, int npad, int k) {
  float inv = 1.0f / (float)(k * k);
  int r = k >> 1;
  for (int i = threadIdx.x; i < npad * npad; i += blockDim.x) {
    int c = i / npad, j = i % npad;
    float v = 0.0f;
    if (c < n && j < n) {
      int mx = c > j ? c : j;
      int mn = c < j ? c : j;
      int lo = mx - r; if (lo < 0) lo = 0;
      int hi = mn + r; if (hi > n - 1) hi = n - 1;
      int cnt = hi - lo + 1; if (cnt < 0) cnt = 0;
      v = (float)cnt * inv;
    }
    M[i] = (_Float16)v;
  }
}

__global__ void k_f32_to_f16(const float* __restrict__ x, _Float16* __restrict__ y, int n) {
  for (int i = blockIdx.x * blockDim.x + threadIdx.x; i < n; i += gridDim.x * blockDim.x)
    y[i] = (_Float16)x[i];
}

// ---------------------------------------------------------------------------
// GEMM A: filter contiguous last axis.  Y[r,j] = sum_c X[r,c] M[c,j].
// A operand: lane row = r, two contiguous 16B segments (K = 8*half+e / +16).
// B operand: M[K][col] == M[col][K] by symmetry -> same contiguous pattern.
// One wave per 16x16 tile; 8 waves per block.
// ---------------------------------------------------------------------------
__global__ void k_gemmA(const _Float16* __restrict__ X, const _Float16* __restrict__ M,
                        _Float16* __restrict__ Y, int P, int N) {
  int wid  = blockIdx.x * (blockDim.x >> 5) + (threadIdx.x >> 5);
  int lane = threadIdx.x & 31;
  int tilesN = N >> 4;
  int tiles  = (P >> 4) * tilesN;
  if (wid >= tiles) return;                       // wave-uniform
  int tr = wid / tilesN, tc = wid % tilesN;
  int m = lane & 15, half = lane >> 4;
  const v8h* xrow = (const v8h*)(X + (size_t)(tr * 16 + m) * N);
  const v8h* mrow = (const v8h*)(M + (size_t)(tc * 16 + m) * N);   // symmetry
  v8f acc = {};
  for (int k0 = 0; k0 < N; k0 += 32) {
    int i0 = (k0 >> 3) + half;
    v16h a = SHUF16(xrow[i0], xrow[i0 + 2]);
    v16h b = SHUF16(mrow[i0], mrow[i0 + 2]);
    acc = __builtin_amdgcn_wmma_f32_16x16x32_f16(false, a, false, b,
                                                 (short)0, acc, false, false);
  }
#pragma unroll
  for (int v = 0; v < 8; ++v)
    Y[(tr * 16 + v + half * 8) * N + tc * 16 + m] = (_Float16)acc[v];
}

// ---------------------------------------------------------------------------
// Stage 32 K-rows x 16 columns of X (stride Q) into an LDS slice [32][16].
// Lane l copies row k0+l (two 16B segments).  Async-to-LDS when available.
// Rows >= N are skipped (guarded at the consumer) / zero-filled (fallback).
// ---------------------------------------------------------------------------
__device__ __forceinline__ void stage32(const _Float16* __restrict__ Xp, int k0,
                                        int N, int Q, _Float16* slice, int lane) {
  int row = k0 + lane;
#if HAVE_ASYNC_LDS
  if (row < N) {
    const _Float16* src = Xp + (size_t)row * Q;
    _Float16* dst = slice + lane * 16;
    __builtin_amdgcn_global_load_async_to_lds_b128(
        (v4i*)src, (lds_v4i_ptr)dst, 0, 0);
    __builtin_amdgcn_global_load_async_to_lds_b128(
        (v4i*)(src + 8), (lds_v4i_ptr)(dst + 8), 0, 0);
  }
#else
  v8h lo = {}, hi = {};
  if (row < N) {
    const v8h* src = (const v8h*)(Xp + (size_t)row * Q);
    lo = src[0];
    hi = src[1];
  }
  v8h* dst = (v8h*)(slice + lane * 16);
  dst[0] = lo;
  dst[1] = hi;
#endif
}

// ---------------------------------------------------------------------------
// GEMM B: filter middle axis, inner stride Q.  Y[p,j,q] = sum_c M[j,c] X[p,c,q].
// A = M rows (contiguous b128, zero-padded to NP).  B = X columns, staged
// through double-buffered per-wave LDS slices.
// ---------------------------------------------------------------------------
__global__ void k_gemmB(const _Float16* __restrict__ X, const _Float16* __restrict__ M,
                        _Float16* __restrict__ Y, int P, int N, int NP, int Q) {
  __shared__ __align__(16) _Float16 stage[8][2][32][16];   // 16 KB
  int w    = threadIdx.x >> 5;
  int lane = threadIdx.x & 31;
  int wid  = blockIdx.x * (blockDim.x >> 5) + w;
  int cq = Q >> 4, cj = NP >> 4;
  int iq = wid % cq;
  int ij = (wid / cq) % cj;
  int p  = wid / (cq * cj);
  if (p >= P) return;                             // wave-uniform (grids exact)
  int m = lane & 15, half = lane >> 4;
  const _Float16* Xp = X + (size_t)p * N * Q + iq * 16;
  const v8h* arow = (const v8h*)(M + (size_t)(ij * 16 + m) * NP);

  stage32(Xp, 0, N, Q, &stage[w][0][0][0], lane);
  v8f acc = {};
  for (int k0 = 0; k0 < NP; k0 += 32) {
    int buf = (k0 >> 5) & 1;
    bool more = (k0 + 32) < NP;
    if (more) stage32(Xp, k0 + 32, N, Q, &stage[w][buf ^ 1][0][0], lane);
#if HAVE_ASYNC_LDS
    if (more) __builtin_amdgcn_s_wait_asynccnt(2);   // current buffer landed
    else      __builtin_amdgcn_s_wait_asynccnt(0);
#endif
    int i0 = (k0 >> 3) + half;
    v16h a = SHUF16(arow[i0], arow[i0 + 2]);
    v16h b;
#pragma unroll
    for (int e = 0; e < 16; ++e) {
      int K = ((e >> 3) << 4) + (half << 3) + (e & 7);
      b[e] = (k0 + K < N) ? stage[w][buf][K][m] : (_Float16)0.0f;
    }
    acc = __builtin_amdgcn_wmma_f32_16x16x32_f16(false, a, false, b,
                                                 (short)0, acc, false, false);
  }
#pragma unroll
  for (int v = 0; v < 8; ++v) {
    int row = ij * 16 + v + half * 8;
    if (row < N) Y[(p * N + row) * Q + iq * 16 + m] = (_Float16)acc[v];
  }
}

// ---------------------------------------------------------------------------
// dice partial sums over pooled f16 fields (b128 vector loads)
// ---------------------------------------------------------------------------
__global__ void k_reduce3(const _Float16* __restrict__ Pv, const _Float16* __restrict__ Tv,
                          float* acc, int n) {
  __shared__ float smem[256];
  const v8h* pv = (const v8h*)Pv;
  const v8h* tv = (const v8h*)Tv;
  int n8 = n >> 3;
  float sp = 0.f, st = 0.f, si = 0.f;
  for (int i = blockIdx.x * blockDim.x + threadIdx.x; i < n8; i += gridDim.x * blockDim.x) {
    v8h pw = pv[i], tw = tv[i];
#pragma unroll
    for (int e = 0; e < 8; ++e) {
      float p = (float)pw[e], t = (float)tw[e];
      sp += p; st += t; si += p * t;
    }
  }
  sp = block_reduce_add(sp, smem);
  st = block_reduce_add(st, smem);
  si = block_reduce_add(si, smem);
  if (threadIdx.x == 0) {
    atomicAdd(&acc[0], sp);
    atomicAdd(&acc[1], st);
    atomicAdd(&acc[2], si);
  }
}

// ---------------------------------------------------------------------------
// temporal monotonicity: sum |d| - d over d = out[:,i+1]-out[:,i]  (b128 loads)
// ---------------------------------------------------------------------------
__global__ void k_mono(const float* __restrict__ out, float* acc) {
  __shared__ float smem[256];
  const int inner4 = INNER >> 2;
  const int n4 = 2 * 9 * inner4;
  float s = 0.0f;
  for (int i = blockIdx.x * blockDim.x + threadIdx.x; i < n4; i += gridDim.x * blockDim.x) {
    int in4 = i % inner4;
    int t   = (i / inner4) % 9;
    int b   = i / (inner4 * 9);
    const float* base = out + (size_t)(b * 10 + t) * INNER + (size_t)in4 * 4;
    __builtin_prefetch(base + 2 * INNER, 0, 0);
    v4f a = *(const v4f*)base;
    v4f c = *(const v4f*)(base + INNER);
#pragma unroll
    for (int e = 0; e < 4; ++e) {
      float d = c[e] - a[e];
      s += fabsf(d) - d;
    }
  }
  s = block_reduce_add(s, smem);
  if (threadIdx.x == 0) atomicAdd(acc, s);
}

// ---------------------------------------------------------------------------
// channel sum of offsets: xs[b,d,h,w] = sum_{c<10,e<3} off[b,c,d,h,w,e]
// ---------------------------------------------------------------------------
__global__ void k_sum_channels(const float* __restrict__ off, float* __restrict__ xs) {
  for (int i = blockIdx.x * blockDim.x + threadIdx.x; i < 2 * INNER;
       i += gridDim.x * blockDim.x) {
    int v = i % INNER;
    int b = i / INNER;
    const float* base = off + (size_t)b * 10 * INNER * 3 + (size_t)v * 3;
    float s = 0.0f;
#pragma unroll
    for (int c = 0; c < 10; ++c) {
      const float* p = base + (size_t)c * INNER * 3;
      s += p[0] + p[1] + p[2];
    }
    xs[i] = s;
  }
}

// ---------------------------------------------------------------------------
// separable 3x3x3 sobel, VALID; order-1 and order-2 sums in one pass.
// ---------------------------------------------------------------------------
__global__ void k_sobel(const float* __restrict__ xs, float* acc) {
  __shared__ float smem[256];
  const int OD = 22, OH = 94, OW = 94;
  const int n = 2 * OD * OH * OW;
  float s1 = 0.0f, s2 = 0.0f;
  for (int i = blockIdx.x * blockDim.x + threadIdx.x; i < n; i += gridDim.x * blockDim.x) {
    int x = i % OW;
    int y = (i / OW) % OH;
    int z = (i / (OW * OH)) % OD;
    int b = i / (OW * OH * OD);
    const float* base = xs + ((size_t)(b * 24 + z) * 96 + y) * 96 + x;
    float gx = 0.f, gy = 0.f, gz = 0.f;
#pragma unroll
    for (int dz = 0; dz < 3; ++dz)
#pragma unroll
      for (int dy = 0; dy < 3; ++dy)
#pragma unroll
        for (int dx = 0; dx < 3; ++dx) {
          float v  = base[(dz * 96 + dy) * 96 + dx];
          float sx = (dx == 1) ? 2.0f : 1.0f;
          float sy = (dy == 1) ? 2.0f : 1.0f;
          gx += v * sy * (float)(1 - dx);
          gy += v * (float)(1 - dy) * sx;
          gz += v * (float)(1 - dz) * sx;
        }
    s1 += gx + gy * gy + gz * gz;
    s2 += sqrtf(gx * gx + gy * gy + gz * gz);
  }
  s1 = block_reduce_add(s1, smem);
  s2 = block_reduce_add(s2, smem);
  if (threadIdx.x == 0) { atomicAdd(&acc[0], s1); atomicAdd(&acc[1], s2); }
}

// ---------------------------------------------------------------------------
// final scalar combine
// ---------------------------------------------------------------------------
__global__ void k_finalize(const float* acc, float* out) {
  if (threadIdx.x != 0 || blockIdx.x != 0) return;
  const float Wp[5] = {0.20f, 0.20f, 0.20f, 0.10f, 0.10f};
  float loss = 0.0f;
  for (int p = 0; p < 5; ++p) {
    float m = 0.0f;
    for (int s = 0; s < 5; ++s) {
      const float* a = acc + (p * 5 + s) * 3;
      m += 1.0f - (2.0f * a[2] + 1e-7f) / (a[0] + a[1] + 1e-7f);
    }
    loss += Wp[p] * (m * 0.25f);
  }
  loss += 0.10f * acc[75] / 442368.0f;
  loss += 0.05f * ((acc[76] + acc[78]) / 388784.0f);
  loss += 0.05f * ((acc[77] + acc[79]) / 388784.0f);
  out[0] = loss;
}

// ---------------------------------------------------------------------------
// host side
// ---------------------------------------------------------------------------
extern "C" void kernel_launch(void* const* d_in, const int* in_sizes, int n_in,
                              void* d_out, int out_size, void* d_ws, size_t ws_size,
                              hipStream_t stream) {
  (void)in_sizes; (void)n_in; (void)out_size;
  const float* vols[10];
  for (int i = 0; i < 10; ++i) vols[i] = (const float*)d_in[i];
  const float* output = (const float*)d_in[10];
  const float* offc   = (const float*)d_in[11];
  const float* offp   = (const float*)d_in[12];
  float* out = (float*)d_out;

  char*  ws  = (char*)d_ws;
  size_t off = 0;
  float*    acc = (float*)(ws + off);      off += 512;
  _Float16* M96 = (_Float16*)(ws + off);   off += (size_t)5 * 96 * 96 * 2;
  _Float16* M24 = (_Float16*)(ws + off);   off += (size_t)5 * 32 * 32 * 2;
  _Float16* X16 = (_Float16*)(ws + off);   off += (size_t)10 * NVOL * 2;
  _Float16* T1  = (_Float16*)(ws + off);   off += (size_t)NVOL * 2;
  _Float16* T2  = (_Float16*)(ws + off);   off += (size_t)NVOL * 2;
  _Float16* PP  = (_Float16*)(ws + off);   off += (size_t)NVOL * 2;
  _Float16* PT  = (_Float16*)(ws + off);   off += (size_t)NVOL * 2;
  float*    XS  = (float*)(ws + off);      off += (size_t)NVOL * 4;

  if (ws_size < off) { k_write_zero<<<1, 1, 0, stream>>>(out); return; }

  k_zero<<<1, 128, 0, stream>>>(acc, 128);

  const int kd[5]  = {1, 3, 5, 7, 9};
  const int khw[5] = {5, 13, 23, 31, 41};
  for (int s = 0; s < 5; ++s) {
    k_build_m<<<1, 256, 0, stream>>>(M96 + (size_t)s * 96 * 96, 96, 96, khw[s]);
    k_build_m<<<1, 256, 0, stream>>>(M24 + (size_t)s * 32 * 32, 24, 32, kd[s]);
  }
  for (int v = 0; v < 10; ++v)
    k_f32_to_f16<<<864, 256, 0, stream>>>(vols[v], X16 + (size_t)v * NVOL, NVOL);

  // 5 pairs x 5 scales: W (gemmA), H (gemmB), D (gemmB), then 3-sum reduce
  for (int pair = 0; pair < 5; ++pair) {
    for (int s = 0; s < 5; ++s) {
      for (int t = 0; t < 2; ++t) {
        const _Float16* src = X16 + (size_t)(2 * pair + t) * NVOL;
        _Float16* dst = t ? PT : PP;
        k_gemmA<<<216, 256, 0, stream>>>(src, M96 + (size_t)s * 9216, T1, 4608, 96);
        k_gemmB<<<216, 256, 0, stream>>>(T1, M96 + (size_t)s * 9216, T2, 48, 96, 96, 96);
        k_gemmB<<<288, 256, 0, stream>>>(T2, M24 + (size_t)s * 1024, dst, 2, 24, 32, 9216);
      }
      k_reduce3<<<432, 256, 0, stream>>>(PP, PT, acc + (pair * 5 + s) * 3, NVOL);
    }
  }

  k_mono<<<972, 256, 0, stream>>>(output, acc + 75);

  k_sum_channels<<<864, 256, 0, stream>>>(offc, XS);
  k_sobel<<<380, 256, 0, stream>>>(XS, acc + 76);
  k_sum_channels<<<864, 256, 0, stream>>>(offp, XS);
  k_sobel<<<380, 256, 0, stream>>>(XS, acc + 78);

  k_finalize<<<1, 1, 0, stream>>>(acc, out);
}
